// AdditiveAttention_1425929142634
// MI455X (gfx1250) — matrix operand order repaired and checked by reading, not compiled, not run on hardware
//
#include <hip/hip_runtime.h>
#include <hip/hip_bf16.h>

#define B_  32
#define T_  2048
#define D_  1024
#define U_  1024
#define NEGBIG (-1.0e9f)

typedef __attribute__((ext_vector_type(16))) __bf16 v16bf;
typedef __attribute__((ext_vector_type(8)))  float  v8f;

union FragBF {
    struct { uint4 lo; uint4 hi; } u;
    v16bf v;
};

__device__ __forceinline__ unsigned short f2bf(float f) {
    unsigned int u = __builtin_bit_cast(unsigned int, f);
    u += 0x7FFFu + ((u >> 16) & 1u);          // round-to-nearest-even
    return (unsigned short)(u >> 16);
}

// ---------------------------------------------------------------------------
// Kernel 1: WvT[n][k] = bf16(Wv[k][n]) via 32x32 LDS tile (coalesced both sides)
// grid (D_/32, U_/32), block (32,8)
// ---------------------------------------------------------------------------
__global__ void wv_transpose_bf16(const float* __restrict__ Wv,
                                  unsigned short* __restrict__ WvT) {
    __shared__ unsigned short tile[32][33];
    int kb = blockIdx.x * 32;
    int nb = blockIdx.y * 32;
    int tx = threadIdx.x, ty = threadIdx.y;
#pragma unroll
    for (int j = 0; j < 4; ++j) {
        int k = ty + j * 8;
        tile[k][tx] = f2bf(Wv[(size_t)(kb + k) * U_ + nb + tx]);
    }
    __syncthreads();
#pragma unroll
    for (int j = 0; j < 4; ++j) {
        int n = ty + j * 8;
        WvT[(size_t)(nb + n) * D_ + kb + tx] = tile[tx][n];
    }
}

// ---------------------------------------------------------------------------
// Kernel 2: qb[b][n] = query[b]·Wq[:,n] + bq[n] + bv[n]   (B_*U_ threads)
// ---------------------------------------------------------------------------
__global__ void qproj_kernel(const float* __restrict__ query,
                             const float* __restrict__ Wq,
                             const float* __restrict__ bq,
                             const float* __restrict__ bv,
                             float* __restrict__ qb) {
    int idx = blockIdx.x * 256 + threadIdx.x;   // [0, B_*U_)
    int b = idx >> 10;
    int n = idx & (U_ - 1);
    float acc = 0.0f;
    for (int k = 0; k < D_; ++k)
        acc += query[b * D_ + k] * Wq[(size_t)k * U_ + n];
    qb[idx] = acc + bq[n] + bv[n];
}

// ---------------------------------------------------------------------------
// Kernel 3: fused  score[b][t] = Vw · tanh(qb[b] + values[b][t]·Wv)  + mask
// One block per (b, 32-row t-tile). 256 threads = 8 waves.
// Wave w covers n in [w*128, w*128+128), both 16-row tiles.
// ---------------------------------------------------------------------------
__global__ void __launch_bounds__(256)
score_gemm_kernel(const float* __restrict__ values,
                  const int* __restrict__ mask,
                  const unsigned short* __restrict__ WvT,
                  const float* __restrict__ qb,
                  const float* __restrict__ Vw,
                  const float* __restrict__ Vb,
                  float* __restrict__ score) {
    __shared__ unsigned short lA[32 * D_];     // 64 KB: 32 t-rows, bf16, k-contig
    __shared__ float sscore[32];

    const int tid  = threadIdx.x;
    const int blk  = blockIdx.x;               // b * 64 + ttile
    const int b    = blk >> 6;
    const int t0   = (blk & 63) * 32;

    // ---- stage 32 rows of values as bf16 into LDS (coalesced float4 loads)
    const float4* src = (const float4*)(values + ((size_t)b * T_ + t0) * D_);
#pragma unroll 4
    for (int i = 0; i < 32; ++i) {             // 8192 float4 / 256 threads
        int idx = tid + i * 256;
        float4 v = src[idx];
        uint2 w;
        w.x = (unsigned)f2bf(v.x) | ((unsigned)f2bf(v.y) << 16);
        w.y = (unsigned)f2bf(v.z) | ((unsigned)f2bf(v.w) << 16);
        *(uint2*)(&lA[idx * 4]) = w;
    }
    if (tid < 32) sscore[tid] = 0.0f;
    __syncthreads();

    const int lane  = tid & 31;
    const int wv    = tid >> 5;                // wave id 0..7
    const int nloc  = lane & 15;
    const int khalf = lane >> 4;               // 0 or 1
    const int nbase = wv * 128;

    v8f acc[2][8];
#pragma unroll
    for (int rt = 0; rt < 2; ++rt)
#pragma unroll
        for (int i = 0; i < 8; ++i)
            acc[rt][i] = (v8f)0.0f;

    // ---- K loop: 32 steps of 16x16x32 bf16 WMMA
    for (int k0 = 0; k0 < D_; k0 += 32) {
        FragBF a[2];
#pragma unroll
        for (int rt = 0; rt < 2; ++rt) {
            int row = rt * 16 + nloc;          // A: lane m = L%16
            const unsigned short* pr = &lA[row * D_ + k0 + khalf * 8];
            a[rt].u.lo = *(const uint4*)(pr);        // elems 0-7 : k0+khalf*8 ..+7
            a[rt].u.hi = *(const uint4*)(pr + 16);   // elems 8-15: k0+16+khalf*8 ..+7
        }
#pragma unroll
        for (int i = 0; i < 8; ++i) {
            int n = nbase + i * 16 + nloc;     // B: lane column = L%16
            FragBF bf;
            const unsigned short* pb = WvT + (size_t)n * D_ + k0 + khalf * 8;
            bf.u.lo = *(const uint4*)(pb);     // elems 0-7: k0+khalf*8..+7, 8-15: k0+16+khalf*8..+7
            bf.u.hi = *(const uint4*)(pb + 16);
            acc[0][i] = __builtin_amdgcn_wmma_f32_16x16x32_bf16(
                false, a[0].v, false, bf.v, (short)0, acc[0][i], false, false);
            acc[1][i] = __builtin_amdgcn_wmma_f32_16x16x32_bf16(
                false, a[1].v, false, bf.v, (short)0, acc[1][i], false, false);
        }
    }

    // ---- epilogue: tanh(acc + qb[n]) * Vw[n], reduce over n
    float p[16];
#pragma unroll
    for (int j = 0; j < 16; ++j) p[j] = 0.0f;
#pragma unroll
    for (int i = 0; i < 8; ++i) {
        int n = nbase + i * 16 + nloc;
        float q  = qb[b * U_ + n];
        float vw = Vw[n];
#pragma unroll
        for (int rt = 0; rt < 2; ++rt)
#pragma unroll
            for (int r = 0; r < 8; ++r)
                p[rt * 8 + r] += tanhf(acc[rt][i][r] + q) * vw;
    }
    // reduce across the 16 lanes of each half (C layout: n = L%16)
#pragma unroll
    for (int off = 1; off < 16; off <<= 1)
#pragma unroll
        for (int j = 0; j < 16; ++j)
            p[j] += __shfl_xor(p[j], off, 32);
    if (nloc == 0) {
#pragma unroll
        for (int rt = 0; rt < 2; ++rt)
#pragma unroll
            for (int r = 0; r < 8; ++r)
                atomicAdd(&sscore[rt * 16 + khalf * 8 + r], p[rt * 8 + r]);
    }
    __syncthreads();

    if (tid < 32) {
        int t = t0 + tid;
        float s = sscore[tid] + Vb[0];
        s += (1.0f - (float)mask[b * T_ + t]) * NEGBIG;
        score[b * T_ + t] = s;
    }
}

// ---------------------------------------------------------------------------
// Kernel 4: softmax over T per batch row -> attention weights
// ---------------------------------------------------------------------------
__global__ void softmax_kernel(const float* __restrict__ score,
                               float* __restrict__ wout) {
    __shared__ float red[256];
    int b = blockIdx.x, tid = threadIdx.x;
    const float* s = score + b * T_;
    float m = -3.4e38f;
    for (int t = tid; t < T_; t += 256) m = fmaxf(m, s[t]);
    red[tid] = m; __syncthreads();
    for (int off = 128; off; off >>= 1) {
        if (tid < off) red[tid] = fmaxf(red[tid], red[tid + off]);
        __syncthreads();
    }
    float mx = red[0]; __syncthreads();
    float sum = 0.0f;
    for (int t = tid; t < T_; t += 256) sum += __expf(s[t] - mx);
    red[tid] = sum; __syncthreads();
    for (int off = 128; off; off >>= 1) {
        if (tid < off) red[tid] += red[tid + off];
        __syncthreads();
    }
    float inv = 1.0f / red[0];
    for (int t = tid; t < T_; t += 256)
        wout[b * T_ + t] = __expf(s[t] - mx) * inv;
}

// ---------------------------------------------------------------------------
// Kernel 5: context partials over t-segments (no atomics)
// grid = B_*4*4 blocks; block handles (b, d-chunk of 256, t-seg of 512)
// ---------------------------------------------------------------------------
__global__ void ctx_partial_kernel(const float* __restrict__ values,
                                   const float* __restrict__ w,
                                   float* __restrict__ part) {
    int blk = blockIdx.x, tid = threadIdx.x;
    int b   = blk >> 4;
    int rem = blk & 15;
    int dch = rem >> 2;
    int ts  = rem & 3;
    int d   = dch * 256 + tid;
    int tb  = ts * 512;
    const float* vb = values + ((size_t)b * T_ + tb) * D_ + d;
    const float* wb = w + b * T_ + tb;
    float acc = 0.0f;
    for (int t = 0; t < 512; ++t)
        acc += wb[t] * vb[(size_t)t * D_];
    part[(size_t)(b * 4 + ts) * D_ + d] = acc;
}

__global__ void ctx_reduce_kernel(const float* __restrict__ part,
                                  float* __restrict__ ctx) {
    int idx = blockIdx.x * 256 + threadIdx.x;  // [0, B_*D_)
    int b = idx >> 10;
    int d = idx & (D_ - 1);
    const float* p = part + (size_t)b * 4 * D_ + d;
    ctx[idx] = p[0] + p[D_] + p[2 * D_] + p[3 * D_];
}

// ---------------------------------------------------------------------------
extern "C" void kernel_launch(void* const* d_in, const int* in_sizes, int n_in,
                              void* d_out, int out_size, void* d_ws, size_t ws_size,
                              hipStream_t stream) {
    const float* query  = (const float*)d_in[0];
    const float* values = (const float*)d_in[1];
    const int*   mask   = (const int*)  d_in[2];
    const float* Wq     = (const float*)d_in[3];
    const float* bq     = (const float*)d_in[4];
    const float* Wv     = (const float*)d_in[5];
    const float* bv     = (const float*)d_in[6];
    const float* Vw     = (const float*)d_in[7];
    const float* Vb     = (const float*)d_in[8];

    float* ctx  = (float*)d_out;               // [B_, D_]
    float* wout = (float*)d_out + B_ * D_;     // [B_, T_]

    char* ws = (char*)d_ws;
    float*          qb    = (float*)(ws);                       // 131072 B
    float*          score = (float*)(ws + 131072);              // 262144 B
    unsigned short* WvT   = (unsigned short*)(ws + 393216);     // 2 MB
    float*          part  = (float*)(ws + 2490368);             // 512 KB

    wv_transpose_bf16<<<dim3(D_ / 32, U_ / 32), dim3(32, 8), 0, stream>>>(Wv, WvT);
    qproj_kernel<<<(B_ * U_) / 256, 256, 0, stream>>>(query, Wq, bq, bv, qb);
    score_gemm_kernel<<<B_ * (T_ / 32), 256, 0, stream>>>(
        values, mask, WvT, qb, Vw, Vb, score);
    softmax_kernel<<<B_, 256, 0, stream>>>(score, wout);
    ctx_partial_kernel<<<B_ * 16, 256, 0, stream>>>(values, wout, part);
    ctx_reduce_kernel<<<(B_ * D_) / 256, 256, 0, stream>>>(part, ctx);
}
